// IntegratingDecomposingNetwork_89747636617305
// MI455X (gfx1250) — compile-verified, matrix operand order checked by reading, not run on hardware
//
#include <hip/hip_runtime.h>

// ---------------------------------------------------------------------------
// IntegratingDecomposingNetwork on MI455X (gfx1250, wave32, WMMA bf16->f32)
// ---------------------------------------------------------------------------

typedef unsigned short u16;
typedef __attribute__((ext_vector_type(8)))  __bf16 v8bf;
typedef __attribute__((ext_vector_type(16))) __bf16 v16bf;
typedef __attribute__((ext_vector_type(8)))  float  v8f;

#define LENGTH   3394
#define IN_DIM   310
#define REP      512
#define MMD_N    2048
#define ROWS     (2 * LENGTH)   // 6788 real rows (b*LENGTH + l)
#define MT       6848           // rows padded to multiple of 64
#define KP_IN    320            // 310 padded to mult of 32 (and 64 for N)
#define LDSROW   40             // 32 halfs + 8 pad (80B) to spread banks

__device__ __forceinline__ u16 f2bf(float f) {
    unsigned int u = __float_as_uint(f);
    unsigned int r = u + 0x7fffu + ((u >> 16) & 1u);   // round-nearest-even
    return (u16)(r >> 16);
}

__device__ __forceinline__ float blockReduce256(float v, float* red) {
    int t = threadIdx.x;
    red[t] = v; __syncthreads();
    for (int off = 128; off > 0; off >>= 1) {
        if (t < off) red[t] += red[t + off];
        __syncthreads();
    }
    return red[0];
}

// ---------------------------------------------------------------------------
// Generic bf16 GEMM:  Y[M x N] = A[M x K] * Wt[N x K]^T + bias   (bias != 0)
// 64x64 output tile / block, 128 threads = 4 waves, each wave 2x2 WMMA frags.
// Next K-tile is prefetched (global_prefetch_b8) while current is staged.
// ---------------------------------------------------------------------------
__global__ __launch_bounds__(128)
void k_gemm_bf16(const u16* __restrict__ A, int lda,
                 const u16* __restrict__ Bt, int ldb,
                 const float* __restrict__ bias,
                 float* __restrict__ Y, int ldy, int K)
{
    __shared__ __attribute__((aligned(16))) u16 As[64 * LDSROW];
    __shared__ __attribute__((aligned(16))) u16 Bs[64 * LDSROW];

    const int rowBase = blockIdx.y * 64;
    const int colBase = blockIdx.x * 64;
    const int t    = threadIdx.x;
    const int wave = t >> 5;
    const int lane = t & 31;
    const int wr   = (wave >> 1) * 32;      // wave row offset in tile
    const int wc   = (wave & 1) * 32;       // wave col offset in tile
    const int hf   = lane >> 4;             // lane half (0/1)
    const int l16  = lane & 15;

    v8f acc[2][2] = {};

    for (int k0 = 0; k0 < K; k0 += 32) {
        // stage 64x32 bf16 tiles of A and Wt via b128 transfers
#pragma unroll
        for (int i = 0; i < 2; ++i) {
            int c   = t + i * 128;          // 0..255 chunk id
            int r   = c >> 2;               // 0..63 tile row
            int seg = c & 3;                // 16B segment in row
            *reinterpret_cast<uint4*>(&As[r * LDSROW + seg * 8]) =
                *reinterpret_cast<const uint4*>(&A[(size_t)(rowBase + r) * lda + k0 + seg * 8]);
            *reinterpret_cast<uint4*>(&Bs[r * LDSROW + seg * 8]) =
                *reinterpret_cast<const uint4*>(&Bt[(size_t)(colBase + r) * ldb + k0 + seg * 8]);
        }
        // prefetch next K-tile into cache while this one is consumed
        if (k0 + 32 < K) {
#pragma unroll
            for (int i = 0; i < 2; ++i) {
                int c = t + i * 128, r = c >> 2, seg = c & 3;
                __builtin_prefetch(&A[(size_t)(rowBase + r) * lda + k0 + 32 + seg * 8], 0, 3);
                __builtin_prefetch(&Bt[(size_t)(colBase + r) * ldb + k0 + 32 + seg * 8], 0, 3);
            }
        }
        __syncthreads();

        // per-lane fragments: K(j) = { 8h..8h+7 , 16+8h..16+8h+7 }
        v16bf afr[2], bfr[2];
#pragma unroll
        for (int ti = 0; ti < 2; ++ti) {
            const u16* pa = &As[(wr + ti * 16 + l16) * LDSROW + hf * 8];
            v8bf alo = *reinterpret_cast<const v8bf*>(pa);
            v8bf ahi = *reinterpret_cast<const v8bf*>(pa + 16);
            afr[ti] = __builtin_shufflevector(alo, ahi, 0,1,2,3,4,5,6,7,8,9,10,11,12,13,14,15);
            const u16* pb = &Bs[(wc + ti * 16 + l16) * LDSROW + hf * 8];
            v8bf blo = *reinterpret_cast<const v8bf*>(pb);
            v8bf bhi = *reinterpret_cast<const v8bf*>(pb + 16);
            bfr[ti] = __builtin_shufflevector(blo, bhi, 0,1,2,3,4,5,6,7,8,9,10,11,12,13,14,15);
        }
#pragma unroll
        for (int ti = 0; ti < 2; ++ti)
#pragma unroll
            for (int tj = 0; tj < 2; ++tj)
                acc[ti][tj] = __builtin_amdgcn_wmma_f32_16x16x32_bf16(
                    false, afr[ti], false, bfr[tj], (short)0, acc[ti][tj], false, false);
        __syncthreads();
    }

#pragma unroll
    for (int ti = 0; ti < 2; ++ti)
#pragma unroll
        for (int tj = 0; tj < 2; ++tj) {
            int col = colBase + wc + tj * 16 + l16;
            float bv = bias[col];
#pragma unroll
            for (int r = 0; r < 8; ++r) {
                int row = rowBase + wr + ti * 16 + hf * 8 + r;
                Y[(size_t)row * ldy + col] = acc[ti][tj][r] + bv;
            }
        }
}

// ---------------------------------------------------------------------------
// MMD gram GEMM: G = T * T^T on bf16 total[4096x512]; fused multi-kernel
// Gaussian sum with signed quadrant accumulation.
// ---------------------------------------------------------------------------
__global__ __launch_bounds__(128)
void k_mmd_gemm(const u16* __restrict__ T,
                const float* __restrict__ sq,
                float* __restrict__ accum)
{
    __shared__ __attribute__((aligned(16))) u16 As[64 * LDSROW];
    __shared__ __attribute__((aligned(16))) u16 Bs[64 * LDSROW];
    __shared__ float red[128];

    const int rowBase = blockIdx.y * 64;
    const int colBase = blockIdx.x * 64;
    const int t = threadIdx.x, wave = t >> 5, lane = t & 31;
    const int wr = (wave >> 1) * 32, wc = (wave & 1) * 32;
    const int hf = lane >> 4, l16 = lane & 15;

    v8f acc[2][2] = {};
    for (int k0 = 0; k0 < REP; k0 += 32) {
#pragma unroll
        for (int i = 0; i < 2; ++i) {
            int c = t + i * 128, r = c >> 2, seg = c & 3;
            *reinterpret_cast<uint4*>(&As[r * LDSROW + seg * 8]) =
                *reinterpret_cast<const uint4*>(&T[(size_t)(rowBase + r) * REP + k0 + seg * 8]);
            *reinterpret_cast<uint4*>(&Bs[r * LDSROW + seg * 8]) =
                *reinterpret_cast<const uint4*>(&T[(size_t)(colBase + r) * REP + k0 + seg * 8]);
        }
        if (k0 + 32 < REP) {
#pragma unroll
            for (int i = 0; i < 2; ++i) {
                int c = t + i * 128, r = c >> 2, seg = c & 3;
                __builtin_prefetch(&T[(size_t)(rowBase + r) * REP + k0 + 32 + seg * 8], 0, 3);
                __builtin_prefetch(&T[(size_t)(colBase + r) * REP + k0 + 32 + seg * 8], 0, 3);
            }
        }
        __syncthreads();
        v16bf afr[2], bfr[2];
#pragma unroll
        for (int ti = 0; ti < 2; ++ti) {
            const u16* pa = &As[(wr + ti * 16 + l16) * LDSROW + hf * 8];
            afr[ti] = __builtin_shufflevector(*reinterpret_cast<const v8bf*>(pa),
                                              *reinterpret_cast<const v8bf*>(pa + 16),
                                              0,1,2,3,4,5,6,7,8,9,10,11,12,13,14,15);
            const u16* pb = &Bs[(wc + ti * 16 + l16) * LDSROW + hf * 8];
            bfr[ti] = __builtin_shufflevector(*reinterpret_cast<const v8bf*>(pb),
                                              *reinterpret_cast<const v8bf*>(pb + 16),
                                              0,1,2,3,4,5,6,7,8,9,10,11,12,13,14,15);
        }
#pragma unroll
        for (int ti = 0; ti < 2; ++ti)
#pragma unroll
            for (int tj = 0; tj < 2; ++tj)
                acc[ti][tj] = __builtin_amdgcn_wmma_f32_16x16x32_bf16(
                    false, afr[ti], false, bfr[tj], (short)0, acc[ti][tj], false, false);
        __syncthreads();
    }

    float local = 0.0f;
#pragma unroll
    for (int ti = 0; ti < 2; ++ti)
#pragma unroll
        for (int tj = 0; tj < 2; ++tj) {
            int j = colBase + wc + tj * 16 + l16;
            float sqj = sq[j];
#pragma unroll
            for (int r = 0; r < 8; ++r) {
                int i = rowBase + wr + ti * 16 + hf * 8 + r;
                float dist = fmaxf(sq[i] + sqj - 2.0f * acc[ti][tj][r], 0.0f);
                float ks = 0.0f, bw = 0.25f;     // SIGMA / MUL^(NUM/2) = 1/4
#pragma unroll
                for (int q = 0; q < 5; ++q) { ks += __expf(-dist / bw); bw *= 2.0f; }
                local += ((i < MMD_N) == (j < MMD_N)) ? ks : -ks;
            }
        }
    red[t] = local; __syncthreads();
    for (int off = 64; off > 0; off >>= 1) {
        if (t < off) red[t] += red[t + off];
        __syncthreads();
    }
    if (t == 0) atomicAdd(accum, red[0]);
}

// ---------------------------------------------------------------------------
// BN (mean/var over batch x feature, per position) + ReLU (+skip), ld = 512
// Compile-time variants: no runtime null-pointer branches in the hot path.
// ---------------------------------------------------------------------------
template<bool HAS_SKIP, bool HAS_F32>
__global__ __launch_bounds__(256)
void k_bn_relu(const float* __restrict__ Y,
               const float* __restrict__ gam, const float* __restrict__ bet,
               const float* __restrict__ skip,
               u16* __restrict__ obf, float* __restrict__ of32)
{
    const int l = blockIdx.x, t = threadIdx.x;
    __shared__ float red[256];
    const size_t b0 = (size_t)l * REP, b1 = (size_t)(LENGTH + l) * REP;
    float v0 = Y[b0 + t], v1 = Y[b0 + t + 256];
    float v2 = Y[b1 + t], v3 = Y[b1 + t + 256];
    float mean = blockReduce256(v0 + v1 + v2 + v3, red) * (1.0f / 1024.0f);
    __syncthreads();
    float d0 = v0 - mean, d1 = v1 - mean, d2 = v2 - mean, d3 = v3 - mean;
    float var = blockReduce256(d0*d0 + d1*d1 + d2*d2 + d3*d3, red) * (1.0f / 1024.0f);
    float rs = rsqrtf(var + 1e-5f);
    float g = gam[l], b = bet[l];
    float o0 = fmaxf(d0 * rs * g + b, 0.0f);
    float o1 = fmaxf(d1 * rs * g + b, 0.0f);
    float o2 = fmaxf(d2 * rs * g + b, 0.0f);
    float o3 = fmaxf(d3 * rs * g + b, 0.0f);
    if (HAS_SKIP) {
        o0 += skip[b0 + t]; o1 += skip[b0 + t + 256];
        o2 += skip[b1 + t]; o3 += skip[b1 + t + 256];
    }
    obf[b0 + t] = f2bf(o0); obf[b0 + t + 256] = f2bf(o1);
    obf[b1 + t] = f2bf(o2); obf[b1 + t + 256] = f2bf(o3);
    if (HAS_F32) {
        of32[b0 + t] = o0; of32[b0 + t + 256] = o1;
        of32[b1 + t] = o2; of32[b1 + t + 256] = o3;
    }
}

// --------------------------- small utility kernels -------------------------
__global__ void k_zero(float* p, int n) {
    int i = blockIdx.x * blockDim.x + threadIdx.x;
    if (i < n) p[i] = 0.0f;
}

// W[K x N] fp32 -> Wt[Np x Kp] bf16 (transposed, zero-padded)
__global__ void k_wprep(const float* __restrict__ w, u16* __restrict__ wt,
                        int K, int N, int Kp, int Np)
{
    int idx = blockIdx.x * 256 + threadIdx.x;
    if (idx >= Np * Kp) return;
    int n = idx / Kp, k = idx % Kp;
    float v = (n < N && k < K) ? w[(size_t)k * N + n] : 0.0f;
    wt[idx] = f2bf(v);
}

__global__ void k_bprep(const float* __restrict__ b, float* __restrict__ bp,
                        int N, int Np)
{
    int i = blockIdx.x * 256 + threadIdx.x;
    if (i < Np) bp[i] = (i < N) ? b[i] : 0.0f;
}

// EEG (ROWS x 310) fp32 -> (MT x 320) bf16 zero-padded
__global__ void k_eeg_prep(const float* __restrict__ eeg, u16* __restrict__ out)
{
    int idx = blockIdx.x * 256 + threadIdx.x;
    if (idx >= MT * KP_IN) return;
    int row = idx / KP_IN, c = idx % KP_IN;
    float v = (row < ROWS && c < IN_DIM) ? eeg[(size_t)row * IN_DIM + c] : 0.0f;
    out[idx] = f2bf(v);
}

// fp32 -> bf16 (+ optional fp32 copy, compile-time)
template<bool COPY>
__global__ void k_cvt_copy(const float* __restrict__ x, u16* __restrict__ bf,
                           float* __restrict__ cpy, int n)
{
    int i = blockIdx.x * 256 + threadIdx.x;
    if (i >= n) return;
    float v = x[i];
    bf[i] = f2bf(v);
    if (COPY) cpy[i] = v;
}

// MSE(Y[:,0:310] vs EEG) accumulated into acc (row per block)
__global__ __launch_bounds__(256)
void k_mse_rec(const float* __restrict__ Y, const float* __restrict__ eeg,
               float* __restrict__ acc)
{
    __shared__ float red[256];
    int row = blockIdx.x, t = threadIdx.x;
    float local = 0.0f;
    for (int c = t; c < IN_DIM; c += 256) {
        float d = Y[(size_t)row * KP_IN + c] - eeg[(size_t)row * IN_DIM + c];
        local += d * d;
    }
    float tot = blockReduce256(local, red);
    if (t == 0) atomicAdd(acc, tot);
}

// fd_rand gather + build integrator-cross input (bf16) + loss_dom accumulate
__global__ __launch_bounds__(256)
void k_gather_dom(const float* __restrict__ comp, const int* __restrict__ perm,
                  u16* __restrict__ zcross, float* __restrict__ accDom)
{
    __shared__ float red[256];
    int row = blockIdx.x, t = threadIdx.x;
    int b = row / LENGTH, l = row % LENGTH;
    int srcrow = b * LENGTH + perm[l];
    float local = 0.0f;
    for (int c = t; c < REP; c += 256) {
        float fd  = comp[(size_t)srcrow * 1024 + c];         // permuted domain
        float fdo = comp[(size_t)row    * 1024 + c];         // original domain
        float fe  = comp[(size_t)row    * 1024 + REP + c];   // emotion
        zcross[(size_t)row * 1024 + c]       = f2bf(fd);
        zcross[(size_t)row * 1024 + REP + c] = f2bf(fe);
        float d = fd - fdo;
        local += d * d;
    }
    float tot = blockReduce256(local, red);
    if (t == 0) atomicAdd(accDom, tot);
}

// gather 2x2048 emotion rows -> bf16 total[4096x512] + row squared norms
__global__ __launch_bounds__(256)
void k_mmd_prep(const float* __restrict__ comp, const int* __restrict__ idx,
                u16* __restrict__ totbf, float* __restrict__ sq)
{
    __shared__ float red[256];
    int i = blockIdx.x, t = threadIdx.x;       // i in 0..4095
    int b = i >> 11;
    int src = b * LENGTH + idx[i & (MMD_N - 1)];
    float local = 0.0f;
    for (int c = t; c < REP; c += 256) {
        float v = comp[(size_t)src * 1024 + REP + c];
        totbf[(size_t)i * REP + c] = f2bf(v);
        local += v * v;
    }
    float tot = blockReduce256(local, red);
    if (t == 0) sq[i] = tot;
}

__global__ __launch_bounds__(256)
void k_copy_feat(const float* __restrict__ comp, float* __restrict__ out)
{
    int row = blockIdx.x, t = threadIdx.x;
    for (int c = t; c < REP; c += 256)
        out[(size_t)row * REP + c] = comp[(size_t)row * 1024 + REP + c];
}

__global__ void k_finalize(const float* __restrict__ acc, float* __restrict__ out)
{
    if (threadIdx.x == 0 && blockIdx.x == 0) {
        out[0] = acc[0] / (float)(ROWS * IN_DIM);          // loss_rec
        out[1] = acc[1] / (float)(ROWS * REP);             // loss_dom
        out[2] = acc[2] / (float)(ROWS * IN_DIM);          // loss_cross
        out[3] = acc[3] / ((float)MMD_N * (float)MMD_N);   // loss_mmd
    }
}

// ---------------------------------------------------------------------------
// Host orchestration
// ---------------------------------------------------------------------------
static void* wsAlloc(char*& p, size_t bytes) {
    void* r = (void*)p;
    p += (bytes + 255) & ~(size_t)255;
    return r;
}

extern "C" void kernel_launch(void* const* d_in, const int* in_sizes, int n_in,
                              void* d_out, int out_size, void* d_ws, size_t ws_size,
                              hipStream_t stream)
{
    (void)in_sizes; (void)n_in; (void)out_size; (void)ws_size;

    // --- inputs (jax pytree flatten: dict keys sorted alphabetically) ---
    const float* EEG  = (const float*)d_in[0];
    const int*   PERM = (const int*)d_in[1];
    const int*   MMDI = (const int*)d_in[2];

    struct Lin { const float *b, *w; };
    struct RB  { const float *bn1_b, *bn1_g, *bn2_b, *bn2_g; Lin l1, l2; };
    int ii = 3;
    auto rdLin = [&](void) { Lin L; L.b = (const float*)d_in[ii++]; L.w = (const float*)d_in[ii++]; return L; };
    auto rdRB  = [&](void) {
        RB r;
        r.bn1_b = (const float*)d_in[ii++]; r.bn1_g = (const float*)d_in[ii++];
        r.bn2_b = (const float*)d_in[ii++]; r.bn2_g = (const float*)d_in[ii++];
        r.l1 = rdLin(); r.l2 = rdLin();
        return r;
    };
    // dec: l0, l3, rb1, rb2   (sorted: l0 < l3 < rb1 < rb2)
    Lin dec_l0 = rdLin(); Lin dec_l3 = rdLin(); RB dec_rb1 = rdRB(); RB dec_rb2 = rdRB();
    // int: same ordering
    Lin int_l0 = rdLin(); Lin int_l3 = rdLin(); RB int_rb1 = rdRB(); RB int_rb2 = rdRB();

    // --- workspace layout ---
    char* wp = (char*)d_ws;
    u16* WT_DEC_L0 = (u16*)wsAlloc(wp, (size_t)512 * KP_IN * 2);
    u16* WT_DEC_L3 = (u16*)wsAlloc(wp, (size_t)1024 * 512 * 2);
    u16* WT_INT_L0 = (u16*)wsAlloc(wp, (size_t)512 * 1024 * 2);
    u16* WT_INT_L3 = (u16*)wsAlloc(wp, (size_t)KP_IN * 512 * 2);
    u16* WT_RB[8];
    for (int i = 0; i < 8; ++i) WT_RB[i] = (u16*)wsAlloc(wp, (size_t)512 * 512 * 2);
    float* B_DEC_L0 = (float*)wsAlloc(wp, 512 * 4);
    float* B_DEC_L3 = (float*)wsAlloc(wp, 1024 * 4);
    float* B_INT_L0 = (float*)wsAlloc(wp, 512 * 4);
    float* B_INT_L3 = (float*)wsAlloc(wp, KP_IN * 4);
    float* B_RB[8];
    for (int i = 0; i < 8; ++i) B_RB[i] = (float*)wsAlloc(wp, 512 * 4);

    float* ACC     = (float*)wsAlloc(wp, 16 * 4);              // 0 rec,1 dom,2 cross,3 mmd
    float* SQ      = (float*)wsAlloc(wp, 4096 * 4);
    u16*   TOTBF   = (u16*)wsAlloc(wp, (size_t)4096 * REP * 2);
    u16*   BF_GEN  = (u16*)wsAlloc(wp, (size_t)MT * 1024 * 2);
    u16*   BF_COMP = (u16*)wsAlloc(wp, (size_t)MT * 1024 * 2);
    u16*   BF_CROSS= (u16*)wsAlloc(wp, (size_t)MT * 1024 * 2);
    float* Yb      = (float*)wsAlloc(wp, (size_t)MT * 1024 * 4);
    float* COMP    = (float*)wsAlloc(wp, (size_t)MT * 1024 * 4);
    float* SKIP_A  = (float*)wsAlloc(wp, (size_t)MT * REP * 4);
    float* SKIP_B  = (float*)wsAlloc(wp, (size_t)MT * REP * 4);

    const int MTB = MT / 64;   // 107 row tiles
    float* outFeat = (float*)d_out;
    float* outLoss = outFeat + (size_t)ROWS * REP;

    k_zero<<<1, 16, 0, stream>>>(ACC, 16);

    // --- weight prep (transpose + bf16) ---
    auto wprep = [&](const float* w, u16* wt, int K, int N, int Kp, int Np) {
        int n = Np * Kp;
        k_wprep<<<(n + 255) / 256, 256, 0, stream>>>(w, wt, K, N, Kp, Np);
    };
    auto bprep = [&](const float* b, float* bp, int N, int Np) {
        k_bprep<<<(Np + 255) / 256, 256, 0, stream>>>(b, bp, N, Np);
    };
    wprep(dec_l0.w, WT_DEC_L0, IN_DIM, 512, KP_IN, 512);  bprep(dec_l0.b, B_DEC_L0, 512, 512);
    wprep(dec_l3.w, WT_DEC_L3, 512, 1024, 512, 1024);     bprep(dec_l3.b, B_DEC_L3, 1024, 1024);
    wprep(int_l0.w, WT_INT_L0, 1024, 512, 1024, 512);     bprep(int_l0.b, B_INT_L0, 512, 512);
    wprep(int_l3.w, WT_INT_L3, 512, IN_DIM, 512, KP_IN);  bprep(int_l3.b, B_INT_L3, IN_DIM, KP_IN);
    const Lin rbl[8] = { dec_rb1.l1, dec_rb1.l2, dec_rb2.l1, dec_rb2.l2,
                         int_rb1.l1, int_rb1.l2, int_rb2.l1, int_rb2.l2 };
    for (int i = 0; i < 8; ++i) {
        wprep(rbl[i].w, WT_RB[i], 512, 512, 512, 512);
        bprep(rbl[i].b, B_RB[i], 512, 512);
    }

    auto gemm = [&](const u16* A, int lda, const u16* Bt, int ldb,
                    const float* bias, float* Y, int N, int K) {
        dim3 g(N / 64, MTB);
        k_gemm_bf16<<<g, 128, 0, stream>>>(A, lda, Bt, ldb, bias, Y, N, K);
    };
    auto resblock = [&](const RB& rb, const u16* wl1, const float* bl1,
                        const u16* wl2, const float* bl2,
                        const float* skipIn, float* skipOut) {
        gemm(BF_GEN, 512, wl1, 512, bl1, Yb, 512, 512);
        k_bn_relu<false, false><<<LENGTH, 256, 0, stream>>>(
            Yb, rb.bn1_g, rb.bn1_b, nullptr, BF_GEN, nullptr);
        gemm(BF_GEN, 512, wl2, 512, bl2, Yb, 512, 512);
        k_bn_relu<true, true><<<LENGTH, 256, 0, stream>>>(
            Yb, rb.bn2_g, rb.bn2_b, skipIn, BF_GEN, skipOut);
    };

    // ================= decomposer =================
    {
        int n = MT * KP_IN;
        k_eeg_prep<<<(n + 255) / 256, 256, 0, stream>>>(EEG, BF_GEN);
    }
    gemm(BF_GEN, KP_IN, WT_DEC_L0, KP_IN, B_DEC_L0, Yb, 512, KP_IN);
    k_cvt_copy<true><<<(MT * 512 + 255) / 256, 256, 0, stream>>>(Yb, BF_GEN, SKIP_A, MT * 512);
    resblock(dec_rb1, WT_RB[0], B_RB[0], WT_RB[1], B_RB[1], SKIP_A, SKIP_B);
    resblock(dec_rb2, WT_RB[2], B_RB[2], WT_RB[3], B_RB[3], SKIP_B, SKIP_A);
    gemm(BF_GEN, 512, WT_DEC_L3, 512, B_DEC_L3, COMP, 1024, 512);
    k_cvt_copy<false><<<(MT * 1024 + 255) / 256, 256, 0, stream>>>(COMP, BF_COMP, nullptr, MT * 1024);

    // ================= integrator (shared for rec / rec_cross) =============
    auto integrator = [&](const u16* Z, float* accSlot) {
        gemm(Z, 1024, WT_INT_L0, 1024, B_INT_L0, Yb, 512, 1024);
        k_cvt_copy<true><<<(MT * 512 + 255) / 256, 256, 0, stream>>>(Yb, BF_GEN, SKIP_A, MT * 512);
        resblock(int_rb1, WT_RB[4], B_RB[4], WT_RB[5], B_RB[5], SKIP_A, SKIP_B);
        resblock(int_rb2, WT_RB[6], B_RB[6], WT_RB[7], B_RB[7], SKIP_B, SKIP_A);
        gemm(BF_GEN, 512, WT_INT_L3, 512, B_INT_L3, Yb, KP_IN, 512);
        k_mse_rec<<<ROWS, 256, 0, stream>>>(Yb, EEG, accSlot);
    };

    integrator(BF_COMP, &ACC[0]);                                 // loss_rec
    k_gather_dom<<<ROWS, 256, 0, stream>>>(COMP, PERM, BF_CROSS, &ACC[1]);  // loss_dom
    integrator(BF_CROSS, &ACC[2]);                                // loss_cross

    // ================= MMD =================
    k_mmd_prep<<<2 * MMD_N, 256, 0, stream>>>(COMP, MMDI, TOTBF, SQ);
    {
        dim3 g((2 * MMD_N) / 64, (2 * MMD_N) / 64);
        k_mmd_gemm<<<g, 128, 0, stream>>>(TOTBF, SQ, &ACC[3]);
    }

    // ================= outputs =================
    k_copy_feat<<<ROWS, 256, 0, stream>>>(COMP, outFeat);
    k_finalize<<<1, 32, 0, stream>>>(ACC, outLoss);
}
